// CEMLP_28046136443388
// MI455X (gfx1250) — matrix operand order repaired and checked by reading, not compiled, not run on hardware
//
#include <hip/hip_runtime.h>
#include <math.h>

#define BSZ 8192
#define CSZ 256
#define NBL 8

typedef __bf16 bf16;
typedef float v8f  __attribute__((ext_vector_type(8)));
typedef bf16  v16bf __attribute__((ext_vector_type(16)));
typedef bf16  v8bf  __attribute__((ext_vector_type(8)));

// ---------------- compile-time geometric-algebra tables ----------------
constexpr int pcnt(int x) { return (x & 1) + ((x >> 1) & 1) + ((x >> 2) & 1); }

struct GaTab {
  int   kk[64];   // output blade index (grade-ordered) for (I,J)
  float sg[64];   // Cayley sign
  int   pp[64];   // path index into gp_w[.,20]
};

constexpr GaTab makeTab() {
  GaTab t{};
  int ORD[8] = {0, 1, 2, 4, 3, 5, 6, 7};  // order idx -> bitmask
  int INV[8] = {0, 1, 2, 4, 3, 5, 6, 7};  // bitmask -> order idx
  int GRD[8] = {0, 1, 1, 1, 2, 2, 2, 3};
  bool valid[4][4][4] = {};
  for (int I = 0; I < 8; ++I)
    for (int J = 0; J < 8; ++J) {
      int a = ORD[I], b = ORD[J];
      valid[GRD[I]][GRD[J]][pcnt(a ^ b)] = true;
    }
  int pidx[4][4][4] = {};
  int rank = 0;
  for (int i = 0; i < 4; ++i)
    for (int j = 0; j < 4; ++j)
      for (int k = 0; k < 4; ++k)
        if (valid[i][j][k]) pidx[i][j][k] = rank++;
  for (int I = 0; I < 8; ++I)
    for (int J = 0; J < 8; ++J) {
      int a = ORD[I], b = ORD[J];
      int x = a ^ b;
      int s = 0, a_ = a >> 1;
      while (a_) { s += pcnt(a_ & b); a_ >>= 1; }
      int e = I * 8 + J;
      t.kk[e] = INV[x];
      t.sg[e] = (s & 1) ? -1.0f : 1.0f;
      t.pp[e] = pidx[GRD[I]][GRD[J]][pcnt(x)];
    }
  return t;
}

constexpr GaTab GT = makeTab();
constexpr int GRADE[8] = {0, 1, 1, 1, 2, 2, 2, 3};

// ---------------- pack kernels (f32 -> bf16, blade/grade major) ----------------
__global__ __launch_bounds__(256) void ga_pack_x(const float* __restrict__ x,
                                                 bf16* __restrict__ A) {
  const int k = blockIdx.y;
  const size_t i = (size_t)blockIdx.x * 256 + threadIdx.x;  // b*C + c
  A[(size_t)k * BSZ * CSZ + i] = (bf16)x[i * NBL + k];
}

__global__ __launch_bounds__(256) void ga_pack_w(const float* __restrict__ w,
                                                 bf16* __restrict__ W) {
  const int g = blockIdx.y;
  const size_t oi = (size_t)blockIdx.x * 256 + threadIdx.x;  // o*C + i
  W[(size_t)g * CSZ * CSZ + oi] = (bf16)w[oi * 4 + g];
}

// ---------------- fused grade-wise WMMA GEMM ----------------
// A: [8][B][C] bf16 blade-major, W: [4][C][C] bf16 grade-major (K contiguous)
// Each block: 128 rows (8 waves x 16) x 16 cols x 8 blades.
// MODE 0: + b_lin, mv_silu            -> outB bf16 [8][B][C]
// MODE 1: grade_norm(norm_a)          -> outF f32  [B][C][8]
// MODE 2: + b_left                    -> outF f32  [B][C][8]
template <int MODE>
__global__ __launch_bounds__(256) void ga_gemm(const bf16* __restrict__ A,
                                               const bf16* __restrict__ W,
                                               const float* __restrict__ bias,
                                               const float* __restrict__ p0,
                                               const float* __restrict__ p1,
                                               bf16* __restrict__ outB,
                                               float* __restrict__ outF) {
  __shared__ __align__(32) bf16 wl[4][16][CSZ];  // 32 KB weight tile (all K)
  const int o0 = blockIdx.x * 16;
  const int b0 = blockIdx.y * 128;
  const int tid = threadIdx.x;
  const int wave = tid >> 5;
  const int lane = tid & 31;
  const int lo = lane & 15;
  const int hi = lane >> 4;

  // cooperative weight staging: wl[g][n][k] = W[g][o0+n][k]
  {
    const int lk = (tid & 31) * 8;  // K chunk start
#pragma unroll
    for (int r = 0; r < 8; ++r) {
      const int gn = r * 8 + (tid >> 5);  // g*16 + n
      const int g = gn >> 4, n = gn & 15;
      const bf16* src = W + (((size_t)g * CSZ) + (o0 + n)) * CSZ + lk;
      *(v8bf*)&wl[g][n][lk] = *(const v8bf*)src;
    }
  }
  __syncthreads();

  const int rowA = b0 + wave * 16 + lo;  // A-fragment row for this lane

  // prefetch the 8 blade rows this wave will stream
#pragma unroll
  for (int kb = 0; kb < NBL; ++kb)
    __builtin_prefetch(A + ((size_t)kb * BSZ + rowA) * CSZ, 0, 0);

  v8f acc[NBL];
#pragma unroll
  for (int kb = 0; kb < NBL; ++kb) acc[kb] = (v8f)0.0f;

  union AV { v16bf v; v8bf h[2]; };

  for (int kk = 0; kk < CSZ; kk += 32) {
    v16bf bfrag[4];
#pragma unroll
    for (int g = 0; g < 4; ++g)
      bfrag[g] = *(const v16bf*)&wl[g][lo][kk + 16 * hi];
#pragma unroll
    for (int kb = 0; kb < NBL; ++kb) {
      const bf16* ap = A + ((size_t)kb * BSZ + rowA) * CSZ;
      AV av;
      av.h[0] = *(const v8bf*)(ap + kk + 16 * hi);
      av.h[1] = *(const v8bf*)(ap + kk + 16 * hi + 8);
      acc[kb] = __builtin_amdgcn_wmma_f32_16x16x32_bf16(
          false, av.v, false, bfrag[GRADE[kb]], (short)0, acc[kb], false, false);
    }
  }

  const int o = o0 + lo;

  if (MODE == 0) {
    const float bl = bias[o];
    float sa[4], sb[4];
#pragma unroll
    for (int g = 0; g < 4; ++g) { sa[g] = p0[o * 4 + g]; sb[g] = p1[o * 4 + g]; }
#pragma unroll
    for (int v = 0; v < 8; ++v) {
      float y[NBL];
#pragma unroll
      for (int kb = 0; kb < NBL; ++kb) y[kb] = acc[kb][v];
      y[0] += bl;
      float inv[4] = {y[0],
                      y[1] * y[1] + y[2] * y[2] + y[3] * y[3],
                      y[4] * y[4] + y[5] * y[5] + y[6] * y[6],
                      y[7] * y[7]};
      float sg[4];
#pragma unroll
      for (int g = 0; g < 4; ++g)
        sg[g] = 1.0f / (1.0f + __expf(-(sa[g] * inv[g] + sb[g])));
      const int row = b0 + wave * 16 + v + 8 * hi;
#pragma unroll
      for (int kb = 0; kb < NBL; ++kb)
        outB[(size_t)kb * BSZ * CSZ + (size_t)row * CSZ + o] =
            (bf16)(sg[GRADE[kb]] * y[kb]);
    }
  } else if (MODE == 1) {
    float tg[4];
#pragma unroll
    for (int g = 0; g < 4; ++g)
      tg[g] = 1.0f / (1.0f + __expf(-p0[o * 4 + g]));
#pragma unroll
    for (int v = 0; v < 8; ++v) {
      float y[NBL];
#pragma unroll
      for (int kb = 0; kb < NBL; ++kb) y[kb] = acc[kb][v];
      float nr[4] = {sqrtf(y[0] * y[0]),
                     sqrtf(y[1] * y[1] + y[2] * y[2] + y[3] * y[3]),
                     sqrtf(y[4] * y[4] + y[5] * y[5] + y[6] * y[6]),
                     sqrtf(y[7] * y[7])};
      float dv[4];
#pragma unroll
      for (int g = 0; g < 4; ++g) dv[g] = tg[g] * (nr[g] - 1.0f) + 1.0f + 1e-6f;
      const int row = b0 + wave * 16 + v + 8 * hi;
      float* dst = outF + ((size_t)row * CSZ + o) * NBL;
      float4 r0 = {y[0] / dv[0], y[1] / dv[1], y[2] / dv[1], y[3] / dv[1]};
      float4 r1 = {y[4] / dv[2], y[5] / dv[2], y[6] / dv[2], y[7] / dv[3]};
      *(float4*)dst = r0;
      *(float4*)(dst + 4) = r1;
    }
  } else {  // MODE 2
    const float bl = bias[o];
#pragma unroll
    for (int v = 0; v < 8; ++v) {
      float y[NBL];
#pragma unroll
      for (int kb = 0; kb < NBL; ++kb) y[kb] = acc[kb][v];
      y[0] += bl;
      const int row = b0 + wave * 16 + v + 8 * hi;
      float* dst = outF + ((size_t)row * CSZ + o) * NBL;
      float4 r0 = {y[0], y[1], y[2], y[3]};
      float4 r1 = {y[4], y[5], y[6], y[7]};
      *(float4*)dst = r0;
      *(float4*)(dst + 4) = r1;
    }
  }
}

// ---------------- fused geometric product + multivector layernorm ----------------
// one block per batch row b; thread n = channel
__global__ __launch_bounds__(256) void ga_gp_ln(const bf16* __restrict__ T,
                                                const float* __restrict__ XR,
                                                const float* __restrict__ LF,
                                                const float* __restrict__ gpw,
                                                const float* __restrict__ lna,
                                                float* __restrict__ outF,
                                                bf16* __restrict__ outB,
                                                int last) {
  __shared__ float red[256];
  const int b = blockIdx.x;
  const int n = threadIdx.x;

  float t[NBL];
#pragma unroll
  for (int k = 0; k < NBL; ++k)
    t[k] = (float)T[(size_t)k * BSZ * CSZ + (size_t)b * CSZ + n];

  const float4* xp = (const float4*)(XR + ((size_t)b * CSZ + n) * NBL);
  const float4* lp = (const float4*)(LF + ((size_t)b * CSZ + n) * NBL);
  float4 x0 = xp[0], x1 = xp[1], l0 = lp[0], l1 = lp[1];
  float xr[NBL] = {x0.x, x0.y, x0.z, x0.w, x1.x, x1.y, x1.z, x1.w};
  float lf[NBL] = {l0.x, l0.y, l0.z, l0.w, l1.x, l1.y, l1.z, l1.w};

  float w20[20];
#pragma unroll
  for (int p = 0; p < 20; ++p) w20[p] = gpw[(size_t)n * 20 + p];

  float y[NBL] = {0, 0, 0, 0, 0, 0, 0, 0};
#pragma unroll
  for (int I = 0; I < 8; ++I)
#pragma unroll
    for (int J = 0; J < 8; ++J) {
      constexpr float IS2 = 0.70710678118654752f;
      (void)IS2;
      const int e = I * 8 + J;
      y[GT.kk[e]] += GT.sg[e] * w20[GT.pp[e]] * t[I] * xr[J];
    }

  float ns = 0.0f;
#pragma unroll
  for (int k = 0; k < NBL; ++k) {
    y[k] = (lf[k] + y[k]) * 0.70710678118654752f;
    ns += y[k] * y[k];
  }

  red[n] = sqrtf(ns);
  __syncthreads();
  for (int s = 128; s > 0; s >>= 1) {
    if (n < s) red[n] += red[n + s];
    __syncthreads();
  }
  const float mean = red[0] * (1.0f / 256.0f);
  const float scale = lna[n] / (mean + 1e-6f);

  if (last) {
    float* dst = outF + ((size_t)b * CSZ + n) * NBL;
    float4 r0 = {y[0] * scale, y[1] * scale, y[2] * scale, y[3] * scale};
    float4 r1 = {y[4] * scale, y[5] * scale, y[6] * scale, y[7] * scale};
    *(float4*)dst = r0;
    *(float4*)(dst + 4) = r1;
  } else {
#pragma unroll
    for (int k = 0; k < NBL; ++k)
      outB[(size_t)k * BSZ * CSZ + (size_t)b * CSZ + n] = (bf16)(y[k] * scale);
  }
}

// ---------------- host-side orchestration ----------------
extern "C" void kernel_launch(void* const* d_in, const int* in_sizes, int n_in,
                              void* d_out, int out_size, void* d_ws, size_t ws_size,
                              hipStream_t stream) {
  const float* x = (const float*)d_in[0];
  float* out = (float*)d_out;
  char* ws = (char*)d_ws;

  const size_t SZ_BF = (size_t)NBL * BSZ * CSZ * sizeof(bf16);   // 32 MiB
  const size_t SZ_F  = (size_t)BSZ * CSZ * NBL * sizeof(float);  // 64 MiB
  const size_t SZ_W  = (size_t)4 * CSZ * CSZ * sizeof(bf16);     // 512 KiB

  bf16*  ABF  = (bf16*)(ws);
  bf16*  TBF  = (bf16*)(ws + SZ_BF);
  float* XR   = (float*)(ws + 2 * SZ_BF);
  float* LEFT = (float*)(ws + 2 * SZ_BF + SZ_F);
  bf16*  WLIN = (bf16*)(ws + 2 * SZ_BF + 2 * SZ_F);
  bf16*  WRT  = (bf16*)(ws + 2 * SZ_BF + 2 * SZ_F + SZ_W);
  bf16*  WLF  = (bf16*)(ws + 2 * SZ_BF + 2 * SZ_F + 2 * SZ_W);

  ga_pack_x<<<dim3(BSZ * CSZ / 256, NBL), 256, 0, stream>>>(x, ABF);

  const dim3 gw(CSZ * CSZ / 256, 4);
  const dim3 gg(CSZ / 16, BSZ / 128);

  for (int l = 0; l < 2; ++l) {
    const int base = 1 + 10 * l;
    const float* w_lin  = (const float*)d_in[base + 0];
    const float* b_lin  = (const float*)d_in[base + 1];
    const float* silu_a = (const float*)d_in[base + 2];
    const float* silu_b = (const float*)d_in[base + 3];
    const float* w_rght = (const float*)d_in[base + 4];
    const float* w_left = (const float*)d_in[base + 5];
    const float* b_left = (const float*)d_in[base + 6];
    const float* norm_a = (const float*)d_in[base + 7];
    const float* gp_w   = (const float*)d_in[base + 8];
    const float* ln_a   = (const float*)d_in[base + 9];

    ga_pack_w<<<gw, 256, 0, stream>>>(w_lin, WLIN);
    ga_pack_w<<<gw, 256, 0, stream>>>(w_rght, WRT);
    ga_pack_w<<<gw, 256, 0, stream>>>(w_left, WLF);

    ga_gemm<0><<<gg, 256, 0, stream>>>(ABF, WLIN, b_lin, silu_a, silu_b, TBF, nullptr);
    ga_gemm<1><<<gg, 256, 0, stream>>>(TBF, WRT, nullptr, norm_a, nullptr, nullptr, XR);
    ga_gemm<2><<<gg, 256, 0, stream>>>(TBF, WLF, b_left, nullptr, nullptr, nullptr, LEFT);

    ga_gp_ln<<<BSZ, 256, 0, stream>>>(TBF, XR, LEFT, gp_w, ln_a, out, ABF, l == 1);
  }
}
